// ProfiledNextCharBlock_63445256896794
// MI455X (gfx1250) — matrix operand order repaired and checked by reading, not compiled, not run on hardware
//
#include <hip/hip_runtime.h>
#include <cstdint>
#include <cstddef>

typedef __bf16 bf16;
typedef float v8f  __attribute__((ext_vector_type(8)));
typedef bf16  v16bf __attribute__((ext_vector_type(16)));
typedef int   v4i   __attribute__((vector_size(16)));

#define D_MODEL 768
#define D_INNER 1536
#define D_PROJ  3120
#define NH      24
#define DSTATE  8
#define DHEAD   64
#define SEQLEN  2048
#define NBATCH  2
#define MTOK    (NBATCH*SEQLEN)   /* 4096 */
#define DFF     3072

// ---------------------------------------------------------------------------
// CDNA5 async global->LDS copy (ASYNCcnt-tracked DMA, no VGPR round trip)
// Builtin prototype (from compiler diagnostic): pointer params are typed as
// pointers to int __attribute__((vector_size(16))).
// ---------------------------------------------------------------------------
#if __has_builtin(__builtin_amdgcn_global_load_async_to_lds_b128)
#define HAVE_ASYNC_LDS 1
typedef __attribute__((address_space(1))) v4i gv4i;
typedef __attribute__((address_space(3))) v4i lv4i;
#else
#define HAVE_ASYNC_LDS 0
#endif

__device__ __forceinline__ void async_copy16(const bf16* __restrict__ g, bf16* l) {
#if HAVE_ASYNC_LDS
  // generic->as(1): same bit pattern; generic->as(3): low 32 bits are the LDS
  // offset (ISA 10.2 aperture mapping), so the uintptr_t truncation is exact.
  __builtin_amdgcn_global_load_async_to_lds_b128(
      (gv4i*)(uintptr_t)(const void*)g,
      (lv4i*)(uintptr_t)(void*)l,
      0, 0);
#else
  *(uint4*)l = *(const uint4*)g;
#endif
}

__device__ __forceinline__ void wait_async_all() {
#if HAVE_ASYNC_LDS
#if __has_builtin(__builtin_amdgcn_s_wait_asynccnt)
  __builtin_amdgcn_s_wait_asynccnt(0);
#else
  asm volatile("s_wait_asynccnt 0" ::: "memory");
#endif
#endif
}

// ---------------------------------------------------------------------------
// Elementwise / helper kernels
// ---------------------------------------------------------------------------

// fp32 W[K][N] -> bf16 WT[N][K]  (weights pre-transposed so GEMM reads both
// operands with the reduction dim contiguous)
__global__ __launch_bounds__(256) void convert_transpose_kernel(
    const float* __restrict__ src, bf16* __restrict__ dst, int K, int N) {
  int idx = blockIdx.x * 256 + threadIdx.x;
  if (idx >= K * N) return;
  int k = idx / N, n = idx % N;
  dst[(size_t)n * K + k] = (bf16)src[idx];
}

// row-wise RMSNorm, bf16 output (feeds WMMA GEMMs)
__global__ __launch_bounds__(256) void rmsnorm_kernel(
    const float* __restrict__ x, const float* __restrict__ w,
    bf16* __restrict__ out, int cols) {
  int row = blockIdx.x;
  const float* xr = x + (size_t)row * cols;
  __shared__ float red[256];
  float s = 0.f;
  for (int c = threadIdx.x; c < cols; c += 256) { float v = xr[c]; s += v * v; }
  red[threadIdx.x] = s;
  __syncthreads();
  for (int o = 128; o > 0; o >>= 1) {
    if (threadIdx.x < o) red[threadIdx.x] += red[threadIdx.x + o];
    __syncthreads();
  }
  float r = rsqrtf(red[0] / (float)cols + 1e-6f);
  for (int c = threadIdx.x; c < cols; c += 256)
    out[(size_t)row * cols + c] = (bf16)(xr[c] * r * w[c]);
}

// causal depthwise conv (K=4) + SiLU on proj[:,1536:3072]; emits fp32 (scan u)
// and bf16 (B/C-projection GEMM operand)
__global__ __launch_bounds__(256) void conv_silu_kernel(
    const float* __restrict__ proj, const float* __restrict__ conv_w,
    float* __restrict__ v, bf16* __restrict__ vbf) {
  int idx = blockIdx.x * 256 + threadIdx.x;
  if (idx >= MTOK * D_INNER) return;
  int m = idx / D_INNER, c = idx % D_INNER;
  int b = m / SEQLEN, t = m % SEQLEN;
  float s = 0.f;
#pragma unroll
  for (int k = 0; k < 4; ++k) {
    int tt = t - 3 + k;
    if (tt >= 0)
      s += proj[(size_t)(b * SEQLEN + tt) * D_PROJ + D_INNER + c] * conv_w[c * 4 + k];
  }
  float sv = s / (1.f + __expf(-s));  // silu
  v[idx] = sv;
  vbf[idx] = (bf16)sv;
}

// per-(token,head): softplus/sigmoid/Schur factor + L2-normalize B,C rows
__global__ __launch_bounds__(256) void params_kernel(
    const float* __restrict__ proj, const float* __restrict__ bc,
    float* __restrict__ Bn, float* __restrict__ Cn,
    float* __restrict__ Aar, float* __restrict__ dtar, float* __restrict__ Sar) {
  int idx = blockIdx.x * 256 + threadIdx.x;
  if (idx >= MTOK * NH) return;
  int m = idx / NH, h = idx % NH;
  float p0 = proj[(size_t)m * D_PROJ + 2 * D_INNER + h * 2 + 0];
  float p1 = proj[(size_t)m * D_PROJ + 2 * D_INNER + h * 2 + 1];
  float A  = (p0 > 20.f) ? p0 : log1pf(__expf(p0));       // softplus
  float dt = 1.f / (1.f + __expf(-p1));                   // sigmoid
  float S  = 1.f / (1.f + dt * dt * A);
  Aar[idx] = A; dtar[idx] = dt; Sar[idx] = S;

  const float* br = bc + (size_t)m * (2 * NH * DSTATE) + h * (2 * DSTATE);
  float sb = 0.f, sc = 0.f;
#pragma unroll
  for (int i = 0; i < DSTATE; ++i) { sb += br[i] * br[i]; sc += br[DSTATE + i] * br[DSTATE + i]; }
  float rb = rsqrtf(sb + 1e-12f), rc = rsqrtf(sc + 1e-12f);
  size_t base = (size_t)idx * DSTATE;
#pragma unroll
  for (int i = 0; i < DSTATE; ++i) {
    Bn[base + i] = br[i] * rb;
    Cn[base + i] = br[DSTATE + i] * rc;
  }
}

// ---------------------------------------------------------------------------
// Sequential LinOSS-IM scan: one 64-thread block per (batch, head).
// Lane p owns column p of the head; z[8]/xs[8] live in registers.
// Per-step operands staged through LDS in 32-step chunks.
// ---------------------------------------------------------------------------
#define CHUNK 32
__global__ __launch_bounds__(64) void scan_kernel(
    const float* __restrict__ Bn, const float* __restrict__ Cn,
    const float* __restrict__ Aar, const float* __restrict__ dtar,
    const float* __restrict__ Sar, const float* __restrict__ v,
    float* __restrict__ y) {
  int bh = blockIdx.x;          // 0..47
  int b = bh / NH, h = bh % NH;
  int p = threadIdx.x;          // 0..63

  __shared__ float sB[CHUNK][DSTATE], sC[CHUNK][DSTATE];
  __shared__ float sU[CHUNK][DHEAD];
  __shared__ float sdt[CHUNK], sA[CHUNK], sS[CHUNK];

  float z[DSTATE] = {}, xs[DSTATE] = {};

  for (int t0 = 0; t0 < SEQLEN; t0 += CHUNK) {
    for (int i = threadIdx.x; i < CHUNK * DSTATE; i += 64) {
      int s = i >> 3, n = i & 7;
      size_t base = ((size_t)(b * SEQLEN + t0 + s) * NH + h) * DSTATE + n;
      sB[s][n] = Bn[base];
      sC[s][n] = Cn[base];
    }
    for (int i = threadIdx.x; i < CHUNK * DHEAD; i += 64) {
      int s = i >> 6, pp = i & 63;
      sU[s][pp] = v[(size_t)(b * SEQLEN + t0 + s) * D_INNER + h * DHEAD + pp];
    }
    if (threadIdx.x < CHUNK) {
      size_t idx = (size_t)(b * SEQLEN + t0 + threadIdx.x) * NH + h;
      sdt[threadIdx.x] = dtar[idx];
      sA[threadIdx.x]  = Aar[idx];
      sS[threadIdx.x]  = Sar[idx];
    }
    __syncthreads();

    for (int s = 0; s < CHUNK; ++s) {
      float dt = sdt[s], Af = sA[s], Sf = sS[s];
      float up = sU[s][p];
      float yv = 0.f;
#pragma unroll
      for (int n = 0; n < DSTATE; ++n) {
        float bu = sB[s][n] * up;
        z[n] = Sf * (z[n] - dt * Af * xs[n] + dt * bu);
        xs[n] += dt * z[n];
        yv += sC[s][n] * xs[n];
      }
      y[(size_t)(b * SEQLEN + t0 + s) * D_INNER + h * DHEAD + p] = yv;
    }
    __syncthreads();
  }
}

// gated = (y + D_skip[h]*u) * silu(gate); bf16 output feeds out-proj GEMM
__global__ __launch_bounds__(256) void gate_skip_kernel(
    const float* __restrict__ proj, const float* __restrict__ y,
    const float* __restrict__ v, const float* __restrict__ Dskip,
    bf16* __restrict__ gated) {
  int idx = blockIdx.x * 256 + threadIdx.x;
  if (idx >= MTOK * D_INNER) return;
  int m = idx / D_INNER, c = idx % D_INNER;
  int h = c / DHEAD;
  float g = proj[(size_t)m * D_PROJ + c];             // gate slice = first 1536 cols
  float gs = g / (1.f + __expf(-g));                  // silu
  gated[idx] = (bf16)((y[idx] + Dskip[h] * v[idx]) * gs);
}

// ---------------------------------------------------------------------------
// bf16 WMMA GEMM: C[M,N] = A[M,K] * B^T  (B stored as N x K, K contiguous).
// Block: 128x64 tile, 256 threads = 8 waves, each wave = 2x2 grid of
// v_wmma_f32_16x16x32_bf16 tiles, BK=64 (2 k-steps per staged tile).
// Tiles staged with GLOBAL_LOAD_ASYNC_TO_LDS_B128 and double-buffered in LDS
// so the async DMA of tile k+1 overlaps the 8 WMMAs of tile k.
// EPI: 0=none 1=bias 2=bias+gelu 3=residual 4=bias+residual
// ---------------------------------------------------------------------------
#define BM 128
#define BN 64
#define BK 64

union Frag { v16bf v; uint4 q[2]; };

template <int EPI>
__global__ __launch_bounds__(256) void gemm_bf16_wmma(
    const bf16* __restrict__ A, const bf16* __restrict__ Bw,
    float* __restrict__ C, bf16* __restrict__ Cbf,
    const float* __restrict__ bias, const float* __restrict__ resid,
    int M, int N, int K) {
  __shared__ alignas(16) bf16 As[2][BM][BK];   // 32 KB
  __shared__ alignas(16) bf16 Bs[2][BN][BK];   // 16 KB

  const int bm = blockIdx.y * BM;
  const int bn = blockIdx.x * BN;
  const int tid  = threadIdx.x;
  const int lane = tid & 31;
  const int wave = tid >> 5;   // 0..7
  const int wm   = wave >> 1;  // 0..3 : 32-row strip
  const int wn   = wave & 1;   // 0..1 : 32-col strip
  const int half = lane >> 4;  // K-half selector per ISA layout
  const int lr   = lane & 15;

  v8f acc[2][2] = {};

  // stage one 128x64 A tile + 64x64 B tile into LDS buffer `buf`
  auto stage = [&](int buf, int k0) {
#pragma unroll
    for (int i = 0; i < (BM * BK / 8) / 256; ++i) {   // 4 chunks/thread
      int c = i * 256 + tid;
      int row = c >> 3, col8 = (c & 7) << 3;          // 8 chunks per row
      async_copy16(A + (size_t)(bm + row) * K + k0 + col8, &As[buf][row][col8]);
    }
#pragma unroll
    for (int i = 0; i < (BN * BK / 8) / 256; ++i) {   // 2 chunks/thread
      int c = i * 256 + tid;
      int row = c >> 3, col8 = (c & 7) << 3;
      int gn = bn + row;
      if (gn < N) {
        async_copy16(Bw + (size_t)gn * K + k0 + col8, &Bs[buf][row][col8]);
      } else {
        uint4 zz; zz.x = 0u; zz.y = 0u; zz.z = 0u; zz.w = 0u;
        *(uint4*)&Bs[buf][row][col8] = zz;
      }
    }
  };

  auto compute = [&](int buf) {
#pragma unroll
    for (int ks = 0; ks < BK / 32; ++ks) {
      Frag af[2], bfw[2];
#pragma unroll
      for (int tm = 0; tm < 2; ++tm) {
        int row = wm * 32 + tm * 16 + lr;
        int kb = ks * 32 + half * 8;     // lanes 0-15: K 0-7/16-23, 16-31: 8-15/24-31
        af[tm].q[0] = *(const uint4*)&As[buf][row][kb];
        af[tm].q[1] = *(const uint4*)&As[buf][row][kb + 16];
      }
#pragma unroll
      for (int tn = 0; tn < 2; ++tn) {
        int row = wn * 32 + tn * 16 + lr;
        int kb = ks * 32 + half * 8;     // lanes 0-15: K 0-7/16-23, 16-31: 8-15/24-31
        bfw[tn].q[0] = *(const uint4*)&Bs[buf][row][kb];
        bfw[tn].q[1] = *(const uint4*)&Bs[buf][row][kb + 16];
      }
#pragma unroll
      for (int tm = 0; tm < 2; ++tm)
#pragma unroll
        for (int tn = 0; tn < 2; ++tn)
          acc[tm][tn] = __builtin_amdgcn_wmma_f32_16x16x32_bf16(
              false, af[tm].v, false, bfw[tn].v, (short)0, acc[tm][tn],
              false, false);
    }
  };

  const int KT = K / BK;
  stage(0, 0);
  for (int kt = 0; kt < KT; ++kt) {
    wait_async_all();       // my async copies into buf kt&1 are done
    __syncthreads();        // everyone's are done; everyone finished buf (kt+1)&1
    if (kt + 1 < KT) stage((kt + 1) & 1, (kt + 1) * BK);  // overlaps compute
    compute(kt & 1);
  }

  // epilogue + store (C layout: vgpr r, lane -> M = r + 8*half, N = lr)
#pragma unroll
  for (int tm = 0; tm < 2; ++tm) {
#pragma unroll
    for (int tn = 0; tn < 2; ++tn) {
      int gn = bn + wn * 32 + tn * 16 + lr;
      if (gn >= N) continue;
#pragma unroll
      for (int r = 0; r < 8; ++r) {
        int gm = bm + wm * 32 + tm * 16 + r + 8 * half;
        float val = acc[tm][tn][r];
        if (EPI == 1 || EPI == 2 || EPI == 4) val += bias[gn];
        if (EPI == 2) {
          float x3 = val * val * val;
          val = 0.5f * val * (1.f + tanhf(0.7978845608028654f * (val + 0.044715f * x3)));
        }
        if (EPI == 3 || EPI == 4) val += resid[(size_t)gm * N + gn];
        size_t o = (size_t)gm * N + gn;
        if (Cbf) Cbf[o] = (bf16)val;
        if (C)   C[o]   = val;
      }
    }
  }
}

// ---------------------------------------------------------------------------
// Host orchestration
// ---------------------------------------------------------------------------
static inline int ceil_div(int a, int b) { return (a + b - 1) / b; }

extern "C" void kernel_launch(void* const* d_in, const int* in_sizes, int n_in,
                              void* d_out, int out_size, void* d_ws, size_t ws_size,
                              hipStream_t stream) {
  (void)in_sizes; (void)n_in; (void)out_size; (void)ws_size;

  const float* x       = (const float*)d_in[0];
  const float* norm1_w = (const float*)d_in[1];
  const float* w_in    = (const float*)d_in[2];
  const float* conv_w  = (const float*)d_in[3];
  const float* w_bc    = (const float*)d_in[4];
  const float* D_skip  = (const float*)d_in[5];
  const float* w_out   = (const float*)d_in[6];
  const float* norm2_w = (const float*)d_in[7];
  const float* ff_w1   = (const float*)d_in[8];
  const float* ff_b1   = (const float*)d_in[9];
  const float* ff_w2   = (const float*)d_in[10];
  const float* ff_b2   = (const float*)d_in[11];
  float* out = (float*)d_out;

  // deterministic workspace carve-up (256B aligned)
  char* ws = (char*)d_ws;
  auto alloc = [&](size_t bytes) -> void* {
    void* p = (void*)ws;
    ws += (bytes + 255) & ~(size_t)255;
    return p;
  };
  bf16*  w_inT  = (bf16*)alloc((size_t)D_PROJ * D_MODEL * 2);
  bf16*  w_bcT  = (bf16*)alloc((size_t)(2 * NH * DSTATE) * D_INNER * 2);
  bf16*  w_outT = (bf16*)alloc((size_t)D_MODEL * D_INNER * 2);
  bf16*  ffw1T  = (bf16*)alloc((size_t)DFF * D_MODEL * 2);
  bf16*  ffw2T  = (bf16*)alloc((size_t)D_MODEL * DFF * 2);
  bf16*  xn     = (bf16*)alloc((size_t)MTOK * D_MODEL * 2);
  float* proj   = (float*)alloc((size_t)MTOK * D_PROJ * 4);
  float* vbuf   = (float*)alloc((size_t)MTOK * D_INNER * 4);
  bf16*  vbf    = (bf16*)alloc((size_t)MTOK * D_INNER * 2);
  float* bc     = (float*)alloc((size_t)MTOK * 2 * NH * DSTATE * 4);
  float* Bn     = (float*)alloc((size_t)MTOK * NH * DSTATE * 4);
  float* Cn     = (float*)alloc((size_t)MTOK * NH * DSTATE * 4);
  float* Aar    = (float*)alloc((size_t)MTOK * NH * 4);
  float* dtar   = (float*)alloc((size_t)MTOK * NH * 4);
  float* Sar    = (float*)alloc((size_t)MTOK * NH * 4);
  float* ybuf   = (float*)alloc((size_t)MTOK * D_INNER * 4);
  bf16*  gated  = (bf16*)alloc((size_t)MTOK * D_INNER * 2);
  float* x1     = (float*)alloc((size_t)MTOK * D_MODEL * 4);
  bf16*  hbf    = (bf16*)alloc((size_t)MTOK * D_MODEL * 2);
  bf16*  ff1bf  = (bf16*)alloc((size_t)MTOK * DFF * 2);

  // weight convert + transpose (fp32 KxN -> bf16 NxK)
  auto ct = [&](const float* src, bf16* dst, int K, int N) {
    convert_transpose_kernel<<<ceil_div(K * N, 256), 256, 0, stream>>>(src, dst, K, N);
  };
  ct(w_in,  w_inT,  D_MODEL, D_PROJ);
  ct(w_bc,  w_bcT,  D_INNER, 2 * NH * DSTATE);
  ct(w_out, w_outT, D_INNER, D_MODEL);
  ct(ff_w1, ffw1T,  D_MODEL, DFF);
  ct(ff_w2, ffw2T,  DFF,     D_MODEL);

  auto ggrid = [&](int M, int N) { return dim3((unsigned)ceil_div(N, BN), (unsigned)(M / BM)); };

  // 1) norm1 -> bf16
  rmsnorm_kernel<<<MTOK, 256, 0, stream>>>(x, norm1_w, xn, D_MODEL);
  // 2) in-projection
  gemm_bf16_wmma<0><<<ggrid(MTOK, D_PROJ), 256, 0, stream>>>(
      xn, w_inT, proj, nullptr, nullptr, nullptr, MTOK, D_PROJ, D_MODEL);
  // 3) causal depthwise conv + silu
  conv_silu_kernel<<<ceil_div(MTOK * D_INNER, 256), 256, 0, stream>>>(proj, conv_w, vbuf, vbf);
  // 4) B/C projection
  gemm_bf16_wmma<0><<<ggrid(MTOK, 2 * NH * DSTATE), 256, 0, stream>>>(
      vbf, w_bcT, bc, nullptr, nullptr, nullptr, MTOK, 2 * NH * DSTATE, D_INNER);
  // 5) params + l2norm
  params_kernel<<<ceil_div(MTOK * NH, 256), 256, 0, stream>>>(proj, bc, Bn, Cn, Aar, dtar, Sar);
  // 6) sequential scan over T
  scan_kernel<<<NBATCH * NH, 64, 0, stream>>>(Bn, Cn, Aar, dtar, Sar, vbuf, ybuf);
  // 7) gate + skip
  gate_skip_kernel<<<ceil_div(MTOK * D_INNER, 256), 256, 0, stream>>>(proj, ybuf, vbuf, D_skip, gated);
  // 8) out-projection + residual -> x1
  gemm_bf16_wmma<3><<<ggrid(MTOK, D_MODEL), 256, 0, stream>>>(
      gated, w_outT, x1, nullptr, nullptr, x, MTOK, D_MODEL, D_INNER);
  // 9) norm2 -> bf16
  rmsnorm_kernel<<<MTOK, 256, 0, stream>>>(x1, norm2_w, hbf, D_MODEL);
  // 10) FFN up + bias + gelu -> bf16
  gemm_bf16_wmma<2><<<ggrid(MTOK, DFF), 256, 0, stream>>>(
      hbf, ffw1T, nullptr, ff1bf, ff_b1, nullptr, MTOK, DFF, D_MODEL);
  // 11) FFN down + bias + residual -> output
  gemm_bf16_wmma<4><<<ggrid(MTOK, D_MODEL), 256, 0, stream>>>(
      ff1bf, ffw2T, out, nullptr, ff_b2, x1, MTOK, D_MODEL, DFF);
}
